// Model_24292335026636
// MI455X (gfx1250) — matrix operand order repaired and checked by reading, not compiled, not run on hardware
//
#include <hip/hip_runtime.h>
#include <hip/hip_bf16.h>

// ---------------------------------------------------------------------------
// DEQ-S4D forward for MI455X (gfx1250, wave32) — WMMA-centric version.
//
//   inj GEMM:            v_wmma_f32_16x16x32_f16, feed_w staged in LDS.
//   S4D scan (per 16-step chunk, per channel e) as three WMMAs:
//     y_local  = T0_e(16x16 Toeplitz, impulse response) @ u_tile
//     h_local  = W_e (32x16, input->chunk-final state)  @ u_tile
//     y_corr   = Mcorr_e(16x32, entry-state->outputs)   @ hvec(32)
//   Carry across 512 chunks: scalar complex FMA chain (16384 threads).
//   Coefficient matrices precomputed once per layer (closed-form Abar^k).
// ---------------------------------------------------------------------------

typedef __attribute__((ext_vector_type(16))) _Float16 v16h;
typedef __attribute__((ext_vector_type(8)))  float    v8f;

#define B_   8
#define L_   8192
#define E_   128
#define N_   16
#define CH16_ 16               // WMMA chunk length
#define G16_  (L_/CH16_)       // 512 chunks
#define NGRP_ (G16_/16)        // 32 groups of 16 chunks (one wave each)
#define DT_  0.01f
#define ROWS_ (B_*L_)          // 65536
#define MATH_ ((size_t)E_*512) // halves per coefficient matrix (128 x 16 x 32)

__device__ __forceinline__ float gelu_tanh(float x) {
    const float k0 = 0.7978845608028654f;       // sqrt(2/pi)
    float x3 = x * x * x;
    float t  = tanhf(k0 * (x + 0.044715f * x3));
    return 0.5f * x * (1.0f + t);
}

// A-fragment loader: 16x32 f16, contiguous K order (element i: K = 16*kh + i)
__device__ __forceinline__ v16h load_afrag(const _Float16* __restrict__ Mrow,
                                           int kh) {
    v16h a;
    #pragma unroll
    for (int i = 0; i < 16; ++i) {
        int kk = kh * 16 + i;
        a[i] = Mrow[kk];
    }
    return a;
}

// ------------------------- inj = tanh((x*ew+eb) @ fw + fb) ------------------
__global__ void __launch_bounds__(256) inj_gemm_wmma(
    const float* __restrict__ x, const float* __restrict__ ew,
    const float* __restrict__ eb, const float* __restrict__ fw,
    const float* __restrict__ fb, float* __restrict__ inj)
{
    __shared__ _Float16 sfw[E_ * E_];           // 32 KB of 320 KB WGP LDS
    const int tid = threadIdx.x;
    for (int i = tid; i < E_ * E_; i += 256) sfw[i] = (_Float16)fw[i];
    __syncthreads();

    const int lane = tid & 31;
    const int wave = tid >> 5;                  // col tile 0..7
    const int rowBase = blockIdx.x * 16;
    const int m  = lane & 15;
    const int kh = lane >> 4;
    const int col = wave * 16 + (lane & 15);

    const float xv = x[rowBase + m];            // rank-1 embed r = x*ew + eb
    v8f acc = {};
    #pragma unroll
    for (int ks = 0; ks < 4; ++ks) {
        v16h af, bf;
        #pragma unroll
        for (int i = 0; i < 16; ++i) {
            int kk = ks * 32 + kh * 16 + i;
            af[i] = (_Float16)(xv * ew[kk] + eb[kk]);
        }
        #pragma unroll
        for (int i = 0; i < 16; ++i) bf[i] = sfw[(ks * 32 + kh * 16 + i) * E_ + col];
        acc = __builtin_amdgcn_wmma_f32_16x16x32_f16(
            false, af, false, bf, (short)0, acc, false, false);
    }
    const float bias = fb[col];
    #pragma unroll
    for (int v = 0; v < 8; ++v)
        inj[(size_t)(rowBase + v + kh * 8) * E_ + col] = tanhf(acc[v] + bias);
}

// ------------------------- per-layer coefficient matrices -------------------
// taps[ly][mat][e][r*32+k] f16; mat 0=T0, 1=Mcorr, 2=Wtop(re), 3=Wbot(im)
__global__ void __launch_bounds__(256) prep_taps(
    const float* __restrict__ Alog, const float* __restrict__ Aim,
    const float* __restrict__ Cre,  const float* __restrict__ Cim,
    _Float16* __restrict__ taps)
{
    const int t = blockIdx.x * 256 + threadIdx.x;
    if (t >= 6 * E_ * 16) return;
    const int r = t & 15, e = (t >> 4) & 127, ly = t >> 11;
    const size_t po = (size_t)ly * E_ * N_ + e * N_;

    float arr[N_], aii[N_], abr[N_], abi[N_], bbr[N_], bbi[N_], cr[N_], ci[N_];
    #pragma unroll
    for (int n = 0; n < N_; ++n) {
        const float ar = -expf(Alog[po + n]);
        const float ai = Aim[po + n];
        const float ea = expf(DT_ * ar);
        arr[n] = ar; aii[n] = ai;
        abr[n] = ea * cosf(DT_ * ai);
        abi[n] = ea * sinf(DT_ * ai);
        const float den = 1.0f / (ar * ar + ai * ai);
        const float nr = abr[n] - 1.0f, ni = abi[n];    // Bbar = (Abar-1)/A
        bbr[n] = (nr * ar + ni * ai) * den;
        bbi[n] = (ni * ar - nr * ai) * den;
        cr[n] = Cre[po + n]; ci[n] = Cim[po + n];
    }
    // impulse response K[l] = 2*Re(sum_n C*Bbar*Abar^l), l = 0..15
    float Kv[16], sr[N_], si[N_];
    #pragma unroll
    for (int n = 0; n < N_; ++n) {
        sr[n] = cr[n] * bbr[n] - ci[n] * bbi[n];
        si[n] = cr[n] * bbi[n] + ci[n] * bbr[n];
    }
    #pragma unroll
    for (int l = 0; l < 16; ++l) {
        float s = 0.0f;
        #pragma unroll
        for (int n = 0; n < N_; ++n) {
            s += sr[n];
            const float tr = sr[n] * abr[n] - si[n] * abi[n];
            const float ti = sr[n] * abi[n] + si[n] * abr[n];
            sr[n] = tr; si[n] = ti;
        }
        Kv[l] = 2.0f * s;
    }
    _Float16* T0 = taps + (((size_t)ly * 4 + 0) * E_ + e) * 512 + r * 32;
    #pragma unroll
    for (int c = 0; c < 32; ++c) T0[c] = (c <= r) ? (_Float16)Kv[r - c] : (_Float16)0.0f;

    // Mcorr row r: hvec = [hr_0..15, hi_0..15] -> y_{r} via 2*Re(C*Abar^{r+1}*h)
    _Float16* Mc = taps + (((size_t)ly * 4 + 1) * E_ + e) * 512 + r * 32;
    #pragma unroll
    for (int n = 0; n < N_; ++n) {
        const float ea  = expf((float)(r + 1) * DT_ * arr[n]);
        const float ang = (float)(r + 1) * DT_ * aii[n];
        const float pr = ea * cosf(ang), pi = ea * sinf(ang);
        Mc[n]      = (_Float16)( 2.0f * (cr[n] * pr - ci[n] * pi));
        Mc[16 + n] = (_Float16)(-2.0f * (cr[n] * pi + ci[n] * pr));
    }
    // W rows (state n = r): h_fin_n = sum_j Abar_n^{15-j} * Bbar_n * u_j
    _Float16* Wt = taps + (((size_t)ly * 4 + 2) * E_ + e) * 512 + r * 32;
    _Float16* Wb = taps + (((size_t)ly * 4 + 3) * E_ + e) * 512 + r * 32;
    #pragma unroll
    for (int j = 0; j < 16; ++j) {
        const float ea  = expf((float)(15 - j) * DT_ * arr[r]);
        const float ang = (float)(15 - j) * DT_ * aii[r];
        const float pr = ea * cosf(ang), pi = ea * sinf(ang);
        Wt[j] = (_Float16)(pr * bbr[r] - pi * bbi[r]);
        Wb[j] = (_Float16)(pr * bbi[r] + pi * bbr[r]);
    }
    #pragma unroll
    for (int j = 16; j < 32; ++j) { Wt[j] = (_Float16)0.0f; Wb[j] = (_Float16)0.0f; }
}

// ------------------------- LayerNorm over E -> f16 --------------------------
// mode: 0 -> LN(inj), 1 -> LN(U + inj), 2 -> LN(U).  One wave32 per row.
__global__ void __launch_bounds__(256) ln_kernel(
    const float* __restrict__ U, const float* __restrict__ inj,
    _Float16* __restrict__ U16,
    const float* __restrict__ g, const float* __restrict__ bta, int mode)
{
    const int row  = blockIdx.x * 8 + (threadIdx.x >> 5);
    const int lane = threadIdx.x & 31;
    const size_t base = (size_t)row * E_;
    const float4* up = (const float4*)(U   + base);
    const float4* ip = (const float4*)(inj + base);

    float4 v;
    if (mode == 0) v = ip[lane];
    else if (mode == 1) {
        float4 a = up[lane], c = ip[lane];
        v = make_float4(a.x + c.x, a.y + c.y, a.z + c.z, a.w + c.w);
    } else v = up[lane];

    float s  = v.x + v.y + v.z + v.w;
    float s2 = v.x * v.x + v.y * v.y + v.z * v.z + v.w * v.w;
    #pragma unroll
    for (int off = 16; off > 0; off >>= 1) {
        s  += __shfl_xor(s,  off, 32);
        s2 += __shfl_xor(s2, off, 32);
    }
    const float mean = s * (1.0f / E_);
    const float var  = s2 * (1.0f / E_) - mean * mean;
    const float rstd = rsqrtf(var + 1e-5f);

    float4 gg = ((const float4*)g)[lane];
    float4 bb = ((const float4*)bta)[lane];
    _Float16* o = U16 + base + lane * 4;
    o[0] = (_Float16)((v.x - mean) * rstd * gg.x + bb.x);
    o[1] = (_Float16)((v.y - mean) * rstd * gg.y + bb.y);
    o[2] = (_Float16)((v.z - mean) * rstd * gg.z + bb.z);
    o[3] = (_Float16)((v.w - mean) * rstd * gg.w + bb.w);
}

// ------------------------- Scan phase A (WMMA) ------------------------------
// Wave = (b, e, group of 16 chunks). B operand = Upad(32x16 f16), rows 16-31 = 0.
//   y_local (+D*u) -> Y,   chunk-final local states -> SH (f16, WMMA-B order).
__global__ void __launch_bounds__(256) ssm_wmmaA(
    const _Float16* __restrict__ U16, const _Float16* __restrict__ taps,
    float* __restrict__ Y, _Float16* __restrict__ SH,
    const float* __restrict__ Dp)
{
    const int wglob = blockIdx.x * 8 + (threadIdx.x >> 5);
    const int lane  = threadIdx.x & 31;
    const int e  = wglob & 127;
    const int gr = (wglob >> 7) & (NGRP_ - 1);
    const int b  = wglob >> 12;
    const int m  = lane & 15;
    const int kh = lane >> 4;
    const int col = lane & 15;                  // chunk column
    const int g0  = gr * 16;

    v16h bu;
    if (kh == 0) {                               // rows 0-15: u[16 steps]
        const size_t ub = ((size_t)b * L_ + (size_t)(g0 + col) * CH16_) * E_ + e;
        #pragma unroll
        for (int i = 0; i < 16; ++i) bu[i] = U16[ub + (size_t)i * E_];
    } else {                                     // rows 16-31: zero pad
        #pragma unroll
        for (int i = 0; i < 16; ++i) bu[i] = (_Float16)0.0f;
    }
    const _Float16* rowT0 = taps + ((size_t)0 * E_ + e) * 512 + m * 32;
    const _Float16* rowWt = taps + ((size_t)2 * E_ + e) * 512 + m * 32;
    const _Float16* rowWb = taps + ((size_t)3 * E_ + e) * 512 + m * 32;

    v8f zero = {};
    v8f accy = __builtin_amdgcn_wmma_f32_16x16x32_f16(
        false, load_afrag(rowT0, kh), false, bu, (short)0, zero, false, false);
    v8f achr = __builtin_amdgcn_wmma_f32_16x16x32_f16(
        false, load_afrag(rowWt, kh), false, bu, (short)0, zero, false, false);
    v8f achi = __builtin_amdgcn_wmma_f32_16x16x32_f16(
        false, load_afrag(rowWb, kh), false, bu, (short)0, zero, false, false);

    const float dsk = Dp[e];
    #pragma unroll
    for (int v = 0; v < 8; ++v) {               // D row = timestep-in-chunk
        const int t = (g0 + col) * CH16_ + v + kh * 8;
        const size_t yi = ((size_t)b * L_ + t) * E_ + e;
        Y[yi] = accy[v] + dsk * (float)U16[yi];
    }
    const size_t sb = ((size_t)(b * E_ + e) * G16_ + (g0 + col)) * 32;
    #pragma unroll
    for (int v = 0; v < 8; ++v) {               // D row = state index n
        const int n = v + kh * 8;
        SH[sb + n]      = (_Float16)achr[v];
        SH[sb + 16 + n] = (_Float16)achi[v];
    }
}

// ------------------------- Scan carry (scalar, in place) --------------------
// Thread (b,e,n): SH[g] = entry state h_{16g-1}; h' = Abar^16*h + local_fin[g].
__global__ void __launch_bounds__(256) ssm_carry(
    _Float16* __restrict__ SH,
    const float* __restrict__ Alog, const float* __restrict__ Aim,
    float* __restrict__ hOut, int writeH)
{
    const int tid = blockIdx.x * 256 + threadIdx.x;     // 16384 threads
    const int n = tid & 15, e = (tid >> 4) & 127, b = tid >> 11;

    const float ar = -expf(Alog[e * N_ + n]);
    const float ai = Aim[e * N_ + n];
    const float ea = expf((float)CH16_ * DT_ * ar);     // Abar^16 closed form
    const float pr = ea * cosf((float)CH16_ * DT_ * ai);
    const float pi = ea * sinf((float)CH16_ * DT_ * ai);

    float hr = 0.0f, hi = 0.0f;
    const size_t base = (size_t)(b * E_ + e) * G16_ * 32;
    for (int g = 0; g < G16_; ++g) {
        const size_t ix = base + (size_t)g * 32;
        if ((g & 15) == 0) __builtin_prefetch(&SH[ix + 16 * 32], 0, 1);
        const float sr = (float)SH[ix + n];
        const float si = (float)SH[ix + 16 + n];
        SH[ix + n]      = (_Float16)hr;                  // entry state, B order
        SH[ix + 16 + n] = (_Float16)hi;
        const float tr = pr * hr - pi * hi + sr;
        const float ti = pi * hr + pr * hi + si;
        hr = tr; hi = ti;
    }
    if (writeH) {                                        // final hT (complex64)
        const size_t o = ((size_t)(b * E_ + e) * N_ + n) * 2;
        hOut[o] = hr; hOut[o + 1] = hi;
    }
}

// ------------------------- Scan phase C (WMMA) + GeLU -----------------------
// y += Mcorr_e @ hvec;  U = gelu(y) (x2 on layer 5).
__global__ void __launch_bounds__(256) ssm_wmmaC(
    const float* __restrict__ Y, const _Float16* __restrict__ SH,
    const _Float16* __restrict__ taps, float* __restrict__ U, int extraGelu)
{
    const int wglob = blockIdx.x * 8 + (threadIdx.x >> 5);
    const int lane  = threadIdx.x & 31;
    const int e  = wglob & 127;
    const int gr = (wglob >> 7) & (NGRP_ - 1);
    const int b  = wglob >> 12;
    const int m  = lane & 15;
    const int kh = lane >> 4;
    const int col = lane & 15;
    const int g0  = gr * 16;

    // B operand: hvec(32) of chunk g0+col; SH stored exactly in B lane order.
    v16h bh;
    const size_t sb = ((size_t)(b * E_ + e) * G16_ + (g0 + col)) * 32 + kh * 16;
    #pragma unroll
    for (int i = 0; i < 16; ++i) bh[i] = SH[sb + i];

    const _Float16* rowMc = taps + ((size_t)1 * E_ + e) * 512 + m * 32;
    v8f zero = {};
    v8f acc = __builtin_amdgcn_wmma_f32_16x16x32_f16(
        false, load_afrag(rowMc, kh), false, bh, (short)0, zero, false, false);

    #pragma unroll
    for (int v = 0; v < 8; ++v) {
        const int t = (g0 + col) * CH16_ + v + kh * 8;
        const size_t yi = ((size_t)b * L_ + t) * E_ + e;
        float val = Y[yi] + acc[v];
        val = gelu_tanh(val);
        if (extraGelu) val = gelu_tanh(val);             // z = gelu(u) final_act
        U[yi] = val;
    }
}

// ------------------------- tail: pooling, loss, head ------------------------
__global__ void zero_acc(float* acc) { if (threadIdx.x == 0) acc[0] = 0.0f; }

__global__ void __launch_bounds__(256) pool_loss(
    const float* __restrict__ Z, const float* __restrict__ Zp,
    float* __restrict__ M, float* __restrict__ acc)
{
    const int tid = blockIdx.x * 256 + threadIdx.x;     // 1024 threads (b,e)
    const int e = tid & 127, b = tid >> 7;
    const size_t base = (size_t)b * L_ * E_ + e;
    float s = 0.0f, l = 0.0f;
    for (int t = 0; t < L_; ++t) {
        const float z = Z[base + (size_t)t * E_];
        const float d = z - Zp[base + (size_t)t * E_];
        s += z; l += d * d;
    }
    M[b * E_ + e] = s * (1.0f / L_);
    __shared__ float red[256];
    red[threadIdx.x] = l; __syncthreads();
    for (int o = 128; o > 0; o >>= 1) {
        if (threadIdx.x < o) red[threadIdx.x] += red[threadIdx.x + o];
        __syncthreads();
    }
    if (threadIdx.x == 0) atomicAdd(acc, red[0]);
}

__global__ void __launch_bounds__(128) head_kernel(
    const float* __restrict__ M, const float* __restrict__ fcw,
    const float* __restrict__ fcb, const float* __restrict__ acc,
    float* __restrict__ out)
{
    const int tid = threadIdx.x;
    if (tid < 80) {                                      // logits [8,10]
        const int b = tid / 10, j = tid % 10;
        float s = fcb[j];
        for (int e = 0; e < E_; ++e) s += M[b * E_ + e] * fcw[e * 10 + j];
        out[tid] = s;
    } else if (tid == 80) {
        out[80] = acc[0] / (float)((size_t)B_ * L_ * E_);
    } else if (tid == 81) {
        ((int*)out)[80 + 1 + B_ * E_ * N_ * 2] = 3;      // K_ITERS (int32 bits)
    }
}

// ---------------------------------------------------------------------------
extern "C" void kernel_launch(void* const* d_in, const int* in_sizes, int n_in,
                              void* d_out, int out_size, void* d_ws, size_t ws_size,
                              hipStream_t stream)
{
    const float* x    = (const float*)d_in[0];
    const float* ew   = (const float*)d_in[1];
    const float* eb   = (const float*)d_in[2];
    const float* fw   = (const float*)d_in[3];
    const float* fb   = (const float*)d_in[4];
    const float* Alog = (const float*)d_in[5];
    const float* Aim  = (const float*)d_in[6];
    const float* Cre  = (const float*)d_in[7];
    const float* Cim  = (const float*)d_in[8];
    const float* Dp   = (const float*)d_in[9];
    const float* lng  = (const float*)d_in[10];
    const float* lnb  = (const float*)d_in[11];
    const float* fcw  = (const float*)d_in[12];
    const float* fcb  = (const float*)d_in[13];
    float* out = (float*)d_out;

    float* ws = (float*)d_ws;
    const size_t TENS = (size_t)B_ * L_ * E_;            // 8,388,608
    const size_t SHN  = (size_t)B_ * E_ * G16_ * 32;     // 16,777,216 halves
    float*    inj  = ws;
    float*    U    = inj + TENS;
    float*    Y    = U + TENS;
    float*    Zp   = Y + TENS;
    _Float16* U16  = (_Float16*)(Zp + TENS);             // TENS halves
    _Float16* SH   = U16 + TENS;                         // SHN halves
    _Float16* taps = SH + SHN;                           // 6*4*MATH_ halves
    float*    M    = (float*)(taps + (size_t)6 * 4 * MATH_);
    float*    acc  = M + B_ * E_;

    // One-time: inj GEMM + per-layer coefficient matrices
    inj_gemm_wmma<<<ROWS_ / 16, 256, 0, stream>>>(x, ew, eb, fw, fb, inj);
    prep_taps<<<(6 * E_ * 16 + 255) / 256, 256, 0, stream>>>(Alog, Aim, Cre, Cim, taps);

    const int sgrid = (B_ * E_ * NGRP_) / 8;             // 4096 blocks, 8 waves
    for (int it = 0; it < 3; ++it) {                     // DEQ fixed-point
        for (int ly = 0; ly < 6; ++ly) {
            const int mode = (ly == 0) ? (it == 0 ? 0 : 1) : 2;
            ln_kernel<<<ROWS_ / 8, 256, 0, stream>>>(
                U, inj, U16, lng + ly * E_, lnb + ly * E_, mode);

            const _Float16* ltaps = taps + (size_t)ly * 4 * MATH_;
            const float* al = Alog + (size_t)ly * E_ * N_;
            const float* ai = Aim  + (size_t)ly * E_ * N_;

            ssm_wmmaA<<<sgrid, 256, 0, stream>>>(U16, ltaps, Y, SH, Dp + ly * E_);
            const int writeH = (it == 2 && ly == 5);
            ssm_carry<<<(B_ * E_ * N_) / 256, 256, 0, stream>>>(
                SH, al, ai, out + 81, writeH);
            ssm_wmmaC<<<sgrid, 256, 0, stream>>>(Y, SH, ltaps, U, ly == 5);
        }
        if (it == 1)                                      // prev = z snapshot
            hipMemcpyAsync(Zp, U, TENS * sizeof(float),
                           hipMemcpyDeviceToDevice, stream);
    }
    zero_acc<<<1, 32, 0, stream>>>(acc);
    pool_loss<<<(B_ * E_) / 256, 256, 0, stream>>>(U, Zp, M, acc);
    head_kernel<<<1, 128, 0, stream>>>(M, fcw, fcb, acc, out);
}
